// TraceableResidualLMStep_66709432041907
// MI455X (gfx1250) — matrix operand order repaired and checked by reading, not compiled, not run on hardware
//
#include <hip/hip_runtime.h>
#include <math.h>
#include <stdint.h>

// ---------------------------------------------------------------------------
// Problem constants (from reference)
// ---------------------------------------------------------------------------
#define LAYERS   8
#define DMODEL   1024
#define NHEADS   16
#define HEADDIM  64
#define FFDIM    4096
#define TCACHE   4096
#define TP1      4097          // cache length after append
#define KV_LEN   4097
#define NCHUNK   8             // attention key chunks per head
#define CHUNK_KEYS 512
#define KTILE    128           // keys staged per block iteration
#define PART_STRIDE 66         // {m, l, o[64]} per (head, chunk)
#define RES_SCALE  0.49497474683058327f   // 1.4/sqrt(8)
#define SOFT_SCALE 0.125f                  // 1/sqrt(64)

// per-cache sizes
#define CACHE_IN_F4   (NHEADS * TCACHE * (HEADDIM/4))   // 1,048,576 float4
#define CACHE_OUT_ELT ((size_t)NHEADS * TP1 * HEADDIM)  // 4,195,328 floats

typedef __attribute__((ext_vector_type(16))) _Float16 v16h;
typedef __attribute__((ext_vector_type(8)))  float    v8f;
typedef __attribute__((ext_vector_type(4)))  float    v4f;   // native vec4

// ---------------------------------------------------------------------------
// WMMA helpers (wave32, gfx1250). D = A(16x32 f16) * B(32x16 f16) + C(f32)
// A as loaded here (16x32, contiguous K): lane L<16 holds row L, halves 0..7 = K 0..7,
//   halves 8..15 = K 8..15; lane L>=16 holds row L-16, K 16..23 / 24..31.
// B as loaded here: lane holds column (lane&15); lanes<16 cover K 0..15,
//   lanes>=16 cover K 16..31.
// C/D layout: lanes 0..15: VGPR r -> M=r, N=lane; lanes 16..31: M=8+r.
// ---------------------------------------------------------------------------
__device__ __forceinline__ v8f wmma_f16(v16h a, v16h b, v8f c) {
  return __builtin_amdgcn_wmma_f32_16x16x32_f16(
      /*neg_a=*/false, a, /*neg_b=*/false, b,
      /*c_mod=*/(short)0, c, /*reuse_a=*/false, /*reuse_b=*/false);
}

// A fragment whose rows come from xsrc[kc..kc+31] (vector broadcast when xsrc
// is shared across lanes, true row when xsrc is a per-lane row pointer).
__device__ __forceinline__ v16h frag_a_rowvec(const float* __restrict__ xsrc,
                                              int kc, int sel) {
  v16h a;
#pragma unroll
  for (int h = 0; h < 8; ++h) a[h]     = (_Float16)xsrc[kc + 2 * sel + h];
#pragma unroll
  for (int h = 0; h < 8; ++h) a[8 + h] = (_Float16)xsrc[kc + 2 * sel + 8 + h];
  return a;
}

// B fragment from a column-slice of a row-major [K, N] weight matrix.
__device__ __forceinline__ v16h frag_b_cols(const float* __restrict__ wp,
                                            size_t ldw) {
  v16h b;
#pragma unroll
  for (int h = 0; h < 16; ++h) b[h] = (_Float16)wp[(size_t)h * ldw];
  return b;
}

// B fragment with every column equal to q[base .. base+15].
__device__ __forceinline__ v16h frag_b_vec(const float* __restrict__ q, int base) {
  v16h b;
#pragma unroll
  for (int h = 0; h < 16; ++h) b[h] = (_Float16)q[base + h];
  return b;
}

// ---------------------------------------------------------------------------
// Async global->LDS B128 (gfx1250, ASYNCcnt-tracked).  GVS addressing:
// LDS[lds_byte_off] = MEM[saddr + voff]  (16 bytes per lane).
// ---------------------------------------------------------------------------
__device__ __forceinline__ void async_ld_b128(uint32_t lds_byte_off,
                                              uint32_t g_byte_off,
                                              const float* base) {
  asm volatile("global_load_async_to_lds_b128 %0, %1, %2"
               :: "v"(lds_byte_off), "v"(g_byte_off),
                  "s"((unsigned long long)(uintptr_t)base)
               : "memory");
}

// ---------------------------------------------------------------------------
// KV-cache copy: [H, 4096, 64] -> [H, 4097, 64] (row 4096 filled later).
// Launched per layer right before that layer's attention so the stores are
// still L2-resident (192 MB L2) when attention reads them back.
// Sources are read exactly once -> non-temporal loads.
// ---------------------------------------------------------------------------
__global__ void copy_cache_kernel(const v4f* __restrict__ src,
                                  v4f* __restrict__ dst) {
  size_t idx = (size_t)blockIdx.x * blockDim.x + threadIdx.x; // < CACHE_IN_F4
  int head = (int)(idx / (TCACHE * (HEADDIM / 4)));
  int r    = (int)(idx % (TCACHE * (HEADDIM / 4)));
  int row  = r >> 4;
  int d4   = r & 15;
  v4f v = __builtin_nontemporal_load(&src[idx]);
  dst[(size_t)head * (TP1 * (HEADDIM / 4)) + (size_t)row * (HEADDIM / 4) + d4] = v;
}

__global__ void copy_vec_kernel(const float* __restrict__ s,
                                float* __restrict__ d, int n) {
  int i = blockIdx.x * blockDim.x + threadIdx.x;
  if (i < n) d[i] = s[i];
}

// ---------------------------------------------------------------------------
// RMSNorm over D=1024. One block of 256 threads, vec4 per thread.
// ---------------------------------------------------------------------------
__global__ void rmsnorm_kernel(const float* __restrict__ x,
                               const float* __restrict__ w,
                               float* __restrict__ y) {
  __shared__ float red[256];
  int tid = threadIdx.x;
  v4f xv = ((const v4f*)x)[tid];
  float s = xv.x * xv.x + xv.y * xv.y + xv.z * xv.z + xv.w * xv.w;
  red[tid] = s;
  __syncthreads();
#pragma unroll
  for (int off = 128; off > 0; off >>= 1) {
    if (tid < off) red[tid] += red[tid + off];
    __syncthreads();
  }
  float inv = rsqrtf(red[0] * (1.0f / (float)DMODEL) + 1e-5f);
  v4f wv = ((const v4f*)w)[tid];
  v4f yv;
  yv.x = xv.x * inv * wv.x;
  yv.y = xv.y * inv * wv.y;
  yv.z = xv.z * inv * wv.z;
  yv.w = xv.w * inv * wv.w;
  ((v4f*)y)[tid] = yv;
}

// ---------------------------------------------------------------------------
// GEMV via broadcast-A WMMA: out[col] = dot(x[0:K], W[:,col]) (+resid*scale).
// Block = 256 threads = 8 waves, each wave owns 16 consecutive columns.
// ---------------------------------------------------------------------------
__global__ void gemv_wmma_kernel(const float* __restrict__ x,
                                 const float* __restrict__ W,
                                 const float* __restrict__ resid,
                                 float* __restrict__ out,
                                 int K, int N, float rscale) {
  __shared__ float xs[FFDIM];   // max K
  int tid = threadIdx.x;
  for (int i = tid; i < K; i += blockDim.x) xs[i] = x[i];
  __syncthreads();

  int lane = tid & 31, wid = tid >> 5;
  bool hi = lane >= 16;
  int selA = hi ? 8 : 0;
  int selB = hi ? 16 : 0;
  int col  = blockIdx.x * 128 + wid * 16 + (lane & 15);

  v8f c = {};
  for (int kc = 0; kc < K; kc += 32) {
    const float* wp = W + (size_t)(kc + selB) * N + col;
    __builtin_prefetch(wp + (size_t)32 * N, 0, 1);  // next k-tile
    v16h a = frag_a_rowvec(xs, kc, selA);
    v16h b = frag_b_cols(wp, (size_t)N);
    c = wmma_f16(a, b, c);
  }
  if (lane < 16 && col < N) {
    float acc = c[0];
    out[col] = resid ? (resid[col] + rscale * acc) : acc;
  }
}

// ---------------------------------------------------------------------------
// Fused gate/up GEMV + SiLU: act[col] = silu(x.Wg[:,col]) * (x.Wu[:,col]).
// ---------------------------------------------------------------------------
__global__ void gateup_wmma_kernel(const float* __restrict__ x,
                                   const float* __restrict__ Wg,
                                   const float* __restrict__ Wu,
                                   float* __restrict__ act) {
  __shared__ float xs[DMODEL];
  int tid = threadIdx.x;
  for (int i = tid; i < DMODEL; i += blockDim.x) xs[i] = x[i];
  __syncthreads();

  int lane = tid & 31, wid = tid >> 5;
  bool hi = lane >= 16;
  int selA = hi ? 8 : 0;
  int selB = hi ? 16 : 0;
  int col  = blockIdx.x * 128 + wid * 16 + (lane & 15);

  v8f cg = {}, cu = {};
  for (int kc = 0; kc < DMODEL; kc += 32) {
    v16h a = frag_a_rowvec(xs, kc, selA);
    const float* gp = Wg + (size_t)(kc + selB) * FFDIM + col;
    const float* up = Wu + (size_t)(kc + selB) * FFDIM + col;
    __builtin_prefetch(gp + (size_t)32 * FFDIM, 0, 1);
    __builtin_prefetch(up + (size_t)32 * FFDIM, 0, 1);
    cg = wmma_f16(a, frag_b_cols(gp, FFDIM), cg);
    cu = wmma_f16(a, frag_b_cols(up, FFDIM), cu);
  }
  if (lane < 16) {
    float g = cg[0], u = cu[0];
    act[col] = (g / (1.0f + __expf(-g))) * u;   // silu(g) * u
  }
}

// ---------------------------------------------------------------------------
// RoPE on q and new k; append new k (roped) and v to output caches, row T.
// ---------------------------------------------------------------------------
__global__ void rope_append_kernel(float* __restrict__ q,
                                   const float* __restrict__ kn,
                                   const float* __restrict__ vn,
                                   float* __restrict__ kcO,
                                   float* __restrict__ vcO,
                                   const int* __restrict__ posPtr) {
  int t = threadIdx.x;               // 512 threads
  int h = t >> 5, j = t & 31;
  float pos  = (float)posPtr[0];
  float freq = __powf(10000.0f, -(float)(2 * j) / (float)HEADDIM);
  float ang  = pos * freq;
  float cs = __cosf(ang), sn = __sinf(ang);

  int base = h * HEADDIM;
  float q1 = q[base + j], q2 = q[base + 32 + j];
  q[base + j]      = q1 * cs - q2 * sn;
  q[base + 32 + j] = q1 * sn + q2 * cs;

  float k1 = kn[base + j], k2 = kn[base + 32 + j];
  size_t rowOff = ((size_t)h * TP1 + TCACHE) * HEADDIM;
  kcO[rowOff + j]      = k1 * cs - k2 * sn;
  kcO[rowOff + 32 + j] = k1 * sn + k2 * cs;
  vcO[rowOff + j]      = vn[base + j];
  vcO[rowOff + 32 + j] = vn[base + 32 + j];
}

// ---------------------------------------------------------------------------
// Flash-style attention partials. grid = (NCHUNK, NHEADS), 256 threads.
// K tiles (128 keys x 64 dims, f32 = 32 KB) are streamed into LDS with
// double-buffered global_load_async_to_lds_b128 (ASYNCcnt), so tile i+1
// is in flight while tile i is WMMA-scored. Each wave scores 16 keys
// (A = K rows from LDS, B = q broadcast), runs an online softmax, and
// accumulates o from V (coalesced float2 per lane from global).
// ---------------------------------------------------------------------------
__global__ void attn_partial_kernel(const float* __restrict__ q,
                                    const float* __restrict__ kc,
                                    const float* __restrict__ vc,
                                    float* __restrict__ part, int kvLen) {
  __shared__ float ksh[2][KTILE * HEADDIM];   // 2 x 32 KB K tiles
  __shared__ float qs[HEADDIM];
  __shared__ float p16[8][16];
  __shared__ float wm[8], wl[8], wo[8][HEADDIM];

  int chunk = blockIdx.x, head = blockIdx.y;
  int tid = threadIdx.x, lane = tid & 31, wid = tid >> 5;
  bool hi = lane >= 16;
  int selA = hi ? 8 : 0;
  int selB = hi ? 16 : 0;
  int m0   = hi ? 8 : 0;

  if (tid < HEADDIM) qs[tid] = q[head * HEADDIM + tid];

  const float* kbase = kc + (size_t)head * TP1 * HEADDIM;
  const float* vbase = vc + (size_t)head * TP1 * HEADDIM;

  uint32_t kshBase[2];
  kshBase[0] = (uint32_t)(uintptr_t)&ksh[0][0];
  kshBase[1] = (uint32_t)(uintptr_t)&ksh[1][0];

  int chunkStart = chunk * CHUNK_KEYS;
  int chunkEnd   = (chunk == NCHUNK - 1) ? kvLen : chunkStart + CHUNK_KEYS;
  int iters      = (chunkEnd - chunkStart + (KTILE - 1)) / KTILE;

  // 8 async B128 loads per thread per tile: 256 thr x 8 x 16 B = 32 KB.
  auto issue_tile = [&](int buf, int tileStart) {
#pragma unroll
    for (int j = 0; j < 8; ++j) {
      int cidx = tid + j * 256;                 // 0..2047 B128 chunks
      int row  = cidx >> 4;                     // local key row 0..127
      int f4   = cidx & 15;                     // 16B chunk within row
      int grow = min(tileStart + row, kvLen - 1);   // clamp; masked later
      uint32_t goff = (uint32_t)grow * (HEADDIM * 4u) + (uint32_t)f4 * 16u;
      uint32_t loff = kshBase[buf] +
                      (uint32_t)(row * (HEADDIM * 4) + f4 * 16);
      async_ld_b128(loff, goff, kbase);
    }
  };

  issue_tile(0, chunkStart);

  float m_run = -INFINITY, l_run = 0.0f, o0 = 0.0f, o1 = 0.0f;

  for (int it = 0; it < iters; ++it) {
    if (it + 1 < iters) {
      issue_tile((it + 1) & 1, chunkStart + (it + 1) * KTILE);
      asm volatile("s_wait_asynccnt 0x8" ::: "memory");  // tile it landed
    } else {
      asm volatile("s_wait_asynccnt 0x0" ::: "memory");
    }
    __syncthreads();                       // publish tile across all waves

    const float* ktile = &ksh[it & 1][0];
    int tileBase = chunkStart + it * KTILE + wid * 16;
    const float* kr = ktile + (size_t)(wid * 16 + (lane & 15)) * HEADDIM;

    v8f c = {};
#pragma unroll
    for (int c0 = 0; c0 < HEADDIM; c0 += 32) {
      v16h a = frag_a_rowvec(kr, c0, selA);   // 16 K-rows (from LDS)
      v16h b = frag_b_vec(qs, c0 + selB);     // q broadcast columns
      c = wmma_f16(a, b, c);
    }

    float pl[8];
    float tmax = -INFINITY;
#pragma unroll
    for (int r = 0; r < 8; ++r) {
      int key = tileBase + m0 + r;
      float s = (key < kvLen) ? c[r] * SOFT_SCALE : -INFINITY;
      pl[r] = s;
      tmax = fmaxf(tmax, s);
    }
#pragma unroll
    for (int off = 16; off > 0; off >>= 1)
      tmax = fmaxf(tmax, __shfl_xor(tmax, off, 32));

    float newm = fmaxf(m_run, tmax);
    float corr = (newm == -INFINITY) ? 1.0f
               : (m_run == -INFINITY) ? 0.0f : __expf(m_run - newm);
#pragma unroll
    for (int r = 0; r < 8; ++r)
      pl[r] = (pl[r] == -INFINITY || newm == -INFINITY)
                  ? 0.0f : __expf(pl[r] - newm);

    if (lane == 0 || lane == 16) {             // D columns are identical
#pragma unroll
      for (int r = 0; r < 8; ++r) p16[wid][m0 + r] = pl[r];
    }
    __syncthreads();

    float psum = 0.0f;
    o0 *= corr; o1 *= corr;
#pragma unroll
    for (int m = 0; m < 16; ++m) {
      float pv = p16[wid][m];
      psum += pv;
      int key = tileBase + m;
      if (key < kvLen && pv != 0.0f) {
        const float* vr = vbase + (size_t)key * HEADDIM;
        o0 += pv * vr[2 * lane];
        o1 += pv * vr[2 * lane + 1];
      }
    }
    l_run = l_run * corr + psum;
    m_run = newm;
    __syncthreads();                        // reads done before buf reuse
  }

  // Combine 8 waves of this block into one chunk partial.
  wo[wid][2 * lane]     = o0;
  wo[wid][2 * lane + 1] = o1;
  if (lane == 0) { wm[wid] = m_run; wl[wid] = l_run; }
  __syncthreads();

  if (tid < HEADDIM) {
    float M = -INFINITY;
#pragma unroll
    for (int w = 0; w < 8; ++w) M = fmaxf(M, wm[w]);
    float L = 0.0f, od = 0.0f;
#pragma unroll
    for (int w = 0; w < 8; ++w) {
      float e = (wm[w] == -INFINITY || M == -INFINITY) ? 0.0f
                                                       : __expf(wm[w] - M);
      L  += e * wl[w];
      od += e * wo[w][tid];
    }
    float* pp = part + ((size_t)head * NCHUNK + chunk) * PART_STRIDE;
    if (tid == 0) { pp[0] = M; pp[1] = L; }
    pp[2 + tid] = od;
  }
}

// Combine 8 chunk partials per head -> attn_o[head*64 + d].
__global__ void attn_combine_kernel(const float* __restrict__ part,
                                    float* __restrict__ attn_o) {
  int head = blockIdx.x, d = threadIdx.x;   // 64 threads
  float M = -INFINITY;
#pragma unroll
  for (int ch = 0; ch < NCHUNK; ++ch)
    M = fmaxf(M, part[((size_t)head * NCHUNK + ch) * PART_STRIDE]);
  float L = 0.0f, od = 0.0f;
#pragma unroll
  for (int ch = 0; ch < NCHUNK; ++ch) {
    const float* pp = part + ((size_t)head * NCHUNK + ch) * PART_STRIDE;
    float e = (pp[0] == -INFINITY) ? 0.0f : __expf(pp[0] - M);
    L  += e * pp[1];
    od += e * pp[2 + d];
  }
  attn_o[head * HEADDIM + d] = od / L;
}

// ---------------------------------------------------------------------------
// Host-side orchestration (graph-capture safe: launches only, on `stream`).
// ---------------------------------------------------------------------------
extern "C" void kernel_launch(void* const* d_in, const int* in_sizes, int n_in,
                              void* d_out, int out_size, void* d_ws,
                              size_t ws_size, hipStream_t stream) {
  const float* embed = (const float*)d_in[0];
  const float* kcache[LAYERS];
  const float* vcache[LAYERS];
  for (int i = 0; i < LAYERS; ++i) {
    kcache[i] = (const float*)d_in[1 + 2 * i];
    vcache[i] = (const float*)d_in[2 + 2 * i];
  }
  const float* ln1   = (const float*)d_in[17];
  const float* ln2   = (const float*)d_in[18];
  const float* normw = (const float*)d_in[19];
  const float* Wq    = (const float*)d_in[20];
  const float* Wk    = (const float*)d_in[21];
  const float* Wv    = (const float*)d_in[22];
  const float* Wo    = (const float*)d_in[23];
  const float* Wg    = (const float*)d_in[24];
  const float* Wu    = (const float*)d_in[25];
  const float* Wd    = (const float*)d_in[26];
  const int*   pos   = (const int*)d_in[27];

  float* out = (float*)d_out;
  float* ws  = (float*)d_ws;

  // workspace layout (floats)
  float* hidden = ws;            // 1024
  float* hnorm  = ws + 1024;     // 1024
  float* qkv    = ws + 2048;     // 3072 (q,k,v)
  float* attn_o = ws + 5120;     // 1024
  float* act    = ws + 6144;     // 4096
  float* part   = ws + 10240;    // 16*8*66 = 8448

  float* outBase = out + DMODEL; // caches start after normed hidden

  copy_vec_kernel<<<4, 256, 0, stream>>>(embed, hidden, DMODEL);

  for (int i = 0; i < LAYERS; ++i) {
    float* kcO = outBase + (size_t)i * 2 * CACHE_OUT_ELT;
    float* vcO = kcO + CACHE_OUT_ELT;
    size_t wOffD  = (size_t)i * DMODEL * DMODEL;
    size_t wOffFF = (size_t)i * DMODEL * FFDIM;

    // Stream this layer's old cache into the output layout now, so the
    // 33 MB of stores are still L2-resident when attention reads them.
    copy_cache_kernel<<<CACHE_IN_F4 / 256, 256, 0, stream>>>(
        (const v4f*)kcache[i], (v4f*)kcO);
    copy_cache_kernel<<<CACHE_IN_F4 / 256, 256, 0, stream>>>(
        (const v4f*)vcache[i], (v4f*)vcO);

    rmsnorm_kernel<<<1, 256, 0, stream>>>(hidden, ln1 + i * DMODEL, hnorm);

    gemv_wmma_kernel<<<DMODEL / 128, 256, 0, stream>>>(
        hnorm, Wq + wOffD, nullptr, qkv, DMODEL, DMODEL, 0.0f);
    gemv_wmma_kernel<<<DMODEL / 128, 256, 0, stream>>>(
        hnorm, Wk + wOffD, nullptr, qkv + 1024, DMODEL, DMODEL, 0.0f);
    gemv_wmma_kernel<<<DMODEL / 128, 256, 0, stream>>>(
        hnorm, Wv + wOffD, nullptr, qkv + 2048, DMODEL, DMODEL, 0.0f);

    rope_append_kernel<<<1, 512, 0, stream>>>(qkv, qkv + 1024, qkv + 2048,
                                              kcO, vcO, pos);

    attn_partial_kernel<<<dim3(NCHUNK, NHEADS), 256, 0, stream>>>(
        qkv, kcO, vcO, part, KV_LEN);
    attn_combine_kernel<<<NHEADS, HEADDIM, 0, stream>>>(part, attn_o);

    gemv_wmma_kernel<<<DMODEL / 128, 256, 0, stream>>>(
        attn_o, Wo + wOffD, hidden, hidden, DMODEL, DMODEL, RES_SCALE);

    rmsnorm_kernel<<<1, 256, 0, stream>>>(hidden, ln2 + i * DMODEL, hnorm);

    gateup_wmma_kernel<<<FFDIM / 128, 256, 0, stream>>>(
        hnorm, Wg + wOffFF, Wu + wOffFF, act);

    gemv_wmma_kernel<<<DMODEL / 128, 256, 0, stream>>>(
        act, Wd + wOffFF, hidden, hidden, FFDIM, DMODEL, RES_SCALE);
  }

  rmsnorm_kernel<<<1, 256, 0, stream>>>(hidden, normw, out);
}
